// GATJKClassifier_90477781057646
// MI455X (gfx1250) — compile-verified, matrix operand order checked by reading, not run on hardware
//
#include <hip/hip_runtime.h>
#include <hip/hip_bf16.h>

// ---------------------------------------------------------------------------
// GATJKClassifier for MI455X (gfx1250, wave32).  All dense contractions use
// v_wmma_f32_16x16x32_bf16 (bf16 inputs / f32 accumulate).  VGPR fragment
// layouts per CDNA5 ISA 7.12.2:
//   A (16x32 bf16): lane l holds row m=l&15; element j maps to
//       k = (j<8 ? j : j+8) + (l<16 ? 0 : 8)
//   B (32x16 bf16): lane l holds col n=l&15 with the same k mapping.
//   C/D (16x16 f32): VGPR r holds row r (lanes 0-15) / row r+8 (lanes 16-31),
//       col = lane&15.
// ---------------------------------------------------------------------------

typedef __attribute__((ext_vector_type(16))) __bf16 v16bf;
typedef __attribute__((ext_vector_type(8)))  float  v8f;

#define NN      16384   // N_NODES
#define NG      8       // N_GRAPHS
#define NE      131072  // N_EDGES
#define HID     64
#define HEADS   8
#define TEXTD   768
#define NVIS    4096
#define NAUD    4096
#define TVIS    32
#define TAUD    48
#define NEG_SLOPE 0.2f
#define GN_EPS  1e-5f

static __device__ __forceinline__ float sigm_(float x) { return 1.f / (1.f + expf(-x)); }

static __device__ __forceinline__ unsigned f2ord(float f) {
    unsigned u = __float_as_uint(f);
    return (u & 0x80000000u) ? ~u : (u | 0x80000000u);
}
static __device__ __forceinline__ float ord2f(unsigned o) {
    unsigned u = (o & 0x80000000u) ? (o & 0x7fffffffu) : ~o;
    return __uint_as_float(u);
}

// ------------------------------ utility kernels ----------------------------

__global__ void zero_f32_kernel(float* p, int n) {
    int i = blockIdx.x * blockDim.x + threadIdx.x;
    if (i < n) p[i] = 0.f;
}
__global__ void zero_u32_kernel(unsigned* p, int n) {
    int i = blockIdx.x * blockDim.x + threadIdx.x;
    if (i < n) p[i] = 0u;
}
__global__ void batch_count_kernel(const int* __restrict__ batch, unsigned* cnt, int n) {
    int i = blockIdx.x * blockDim.x + threadIdx.x;
    if (i < n) atomicAdd(&cnt[batch[i]], 1u);
}
__global__ void bias_add_kernel(float* __restrict__ x, const float* __restrict__ b,
                                int total, int D) {
    int i = blockIdx.x * blockDim.x + threadIdx.x;
    if (i < total) x[i] += b[i % D];
}

// ------------------------------ WMMA GEMM ----------------------------------
// C[M,N] = A[M,K] @ W[N,K]^T + bias[N]   (row-major, f32 in/out, bf16 compute)
// One 16x16 C tile per wave32.  M,N multiples of 16; K multiple of 32.

__global__ __launch_bounds__(256)
void gemm_wmma_bias(const float* __restrict__ A, const float* __restrict__ W,
                    const float* __restrict__ bias, float* __restrict__ C,
                    int M, int N, int K)
{
    int gtid = blockIdx.x * blockDim.x + threadIdx.x;
    int wave = gtid >> 5;
    int lane = threadIdx.x & 31;
    int ntn  = N >> 4;
    int tiles = (M >> 4) * ntn;
    if (wave >= tiles) return;
    int tile_m = wave / ntn;
    int tile_n = wave - tile_m * ntn;
    int mrow = tile_m * 16 + (lane & 15);
    int ncol = tile_n * 16 + (lane & 15);
    const float* __restrict__ arow = A + (size_t)mrow * K;
    const float* __restrict__ wrow = W + (size_t)ncol * K;
    const int off = (lane < 16) ? 0 : 8;
    v8f acc = {0.f, 0.f, 0.f, 0.f, 0.f, 0.f, 0.f, 0.f};
    for (int kk = 0; kk < K; kk += 32) {
        if (kk + 32 < K) {
            __builtin_prefetch(arow + kk + 32 + off, 0, 1);   // global_prefetch_b8
            __builtin_prefetch(wrow + kk + 32 + off, 0, 1);
        }
        v16bf av, bv;
#pragma unroll
        for (int h = 0; h < 2; ++h) {
            int k0 = kk + off + h * 16;
#pragma unroll
            for (int j = 0; j < 8; ++j) {
                av[h * 8 + j] = (__bf16)arow[k0 + j];
                bv[h * 8 + j] = (__bf16)wrow[k0 + j];
            }
        }
        acc = __builtin_amdgcn_wmma_f32_16x16x32_bf16(false, av, false, bv,
                                                      (short)0, acc, false, false);
    }
#pragma unroll
    for (int r = 0; r < 8; ++r) {
        int mr = tile_m * 16 + r + ((lane < 16) ? 0 : 8);
        C[(size_t)mr * N + ncol] = acc[r] + bias[ncol];
    }
}

// ------------------------- LSTM step (fused WMMA) --------------------------
// gates[B,256] = [x_t | h_prev] @ [W_ih | W_hh]^T + b.   K = in_d + 64.
// reverse=1 reads x at t_eff = len-1-t (packed-sequence backward direction).

__global__ __launch_bounds__(256)
void lstm_gates_wmma(const float* __restrict__ x, int T, int in_d,
                     const int* __restrict__ lens, int t,
                     const float* __restrict__ h,
                     const float* __restrict__ W_ih, const float* __restrict__ W_hh,
                     const float* __restrict__ bias, float* __restrict__ gates,
                     int B, int reverse)
{
    const int NGATE = 256;
    int gtid = blockIdx.x * blockDim.x + threadIdx.x;
    int wave = gtid >> 5;
    int lane = threadIdx.x & 31;
    int ntn  = NGATE >> 4;                   // 16
    int tiles = (B >> 4) * ntn;
    if (wave >= tiles) return;
    int tile_m = wave / ntn;
    int tile_n = wave - tile_m * ntn;
    int m = tile_m * 16 + (lane & 15);
    int n = tile_n * 16 + (lane & 15);
    int len = lens[m];
    int t_eff = reverse ? ((t < len) ? (len - 1 - t) : t) : t;
    const float* __restrict__ xrow = x + ((size_t)m * T + t_eff) * in_d;
    const float* __restrict__ wih  = W_ih + (size_t)n * in_d;
    const float* __restrict__ whh  = W_hh + (size_t)n * 64;
    const int K = in_d + 64;
    const int off = (lane < 16) ? 0 : 8;
    v8f acc = {0.f, 0.f, 0.f, 0.f, 0.f, 0.f, 0.f, 0.f};
    for (int kk = 0; kk < K; kk += 32) {
        v16bf av, bv;
#pragma unroll
        for (int hh = 0; hh < 2; ++hh) {
            int k0 = kk + off + hh * 16;
            // 8-element groups never straddle the in_d boundary (all mult of 8)
            if (k0 < in_d) {
#pragma unroll
                for (int j = 0; j < 8; ++j) {
                    av[hh * 8 + j] = (__bf16)xrow[k0 + j];
                    bv[hh * 8 + j] = (__bf16)wih[k0 + j];
                }
            } else {
                int kh = k0 - in_d;
#pragma unroll
                for (int j = 0; j < 8; ++j) {
                    float hv = h ? h[(size_t)m * 64 + kh + j] : 0.f;
                    av[hh * 8 + j] = (__bf16)hv;
                    bv[hh * 8 + j] = (__bf16)whh[kh + j];
                }
            }
        }
        acc = __builtin_amdgcn_wmma_f32_16x16x32_bf16(false, av, false, bv,
                                                      (short)0, acc, false, false);
    }
#pragma unroll
    for (int r = 0; r < 8; ++r) {
        int mr = tile_m * 16 + r + ((lane < 16) ? 0 : 8);
        gates[(size_t)mr * NGATE + n] = acc[r] + bias[n];
    }
}

// Pointwise state update; freezes h,c past the valid length, emits y = h2*m.
__global__ void lstm_update(const float* __restrict__ gates,
                            const float* __restrict__ c_prev,
                            const float* __restrict__ h_prev,
                            const int* __restrict__ lens, int t,
                            float* __restrict__ h, float* __restrict__ c,
                            float* __restrict__ ycat, int T, int dir, int B)
{
    int idx = blockIdx.x * blockDim.x + threadIdx.x;
    if (idx >= B * 64) return;
    int bi = idx >> 6, ch = idx & 63;
    const float* g = gates + (size_t)bi * 256;
    float gi = g[ch], gf = g[64 + ch], gg = g[128 + ch], go = g[192 + ch];
    float cp = c_prev ? c_prev[idx] : 0.f;
    float hp = h_prev ? h_prev[idx] : 0.f;
    float c2 = sigm_(gf) * cp + sigm_(gi) * tanhf(gg);
    float h2 = sigm_(go) * tanhf(c2);
    int len = lens[bi];
    bool valid = (t < len);
    h[idx] = valid ? h2 : hp;
    c[idx] = valid ? c2 : cp;
    if (ycat) {
        int tout = dir ? (valid ? (len - 1 - t) : t) : t;   // un-reverse bwd output
        ycat[((size_t)bi * T + tout) * 128 + dir * 64 + ch] = valid ? h2 : 0.f;
    }
}

__global__ void concat_h_kernel(const float* __restrict__ hf, const float* __restrict__ hb,
                                float* __restrict__ hcat, int B)
{
    int idx = blockIdx.x * blockDim.x + threadIdx.x;
    if (idx >= B * 128) return;
    int bi = idx >> 7, ch = idx & 127;
    hcat[idx] = (ch < 64) ? hf[bi * 64 + ch] : hb[bi * 64 + (ch - 64)];
}

// ------------------------ modality fusion (scatter) ------------------------

__global__ void mod_scatter_kernel(const float* __restrict__ logits,
                                   const int* __restrict__ node_map,
                                   float* msum, float* mcnt, int B)
{
    int idx = blockIdx.x * blockDim.x + threadIdx.x;
    if (idx >= B * 64) return;
    int i = idx >> 6, ch = idx & 63;
    int j = node_map[i];
    atomicAdd(&msum[(size_t)j * 64 + ch], logits[idx]);
    if (ch == 0) atomicAdd(&mcnt[j], 1.f);
}

// final_x[idx] = segment_mean(h_win) * mean(xp[idx, :])
__global__ void mod_apply_kernel(const float* __restrict__ msum,
                                 const float* __restrict__ mcnt,
                                 const int* __restrict__ indices,
                                 float* __restrict__ fx)
{
    __shared__ float sb[64];
    int j = blockIdx.x, tx = threadIdx.x;
    int node = indices[j];
    float v = fx[(size_t)node * 64 + tx];
    sb[tx] = v;
    __syncthreads();
    for (int s = 32; s > 0; s >>= 1) {
        if (tx < s) sb[tx] += sb[tx + s];
        __syncthreads();
    }
    float mw = sb[0] * (1.f / 64.f);
    float cf = mcnt[j];
    if (cf < 1.f) cf = 1.f;
    fx[(size_t)node * 64 + tx] = (msum[(size_t)j * 64 + tx] / cf) * mw;
}

// ------------------------------ GATv2 edges --------------------------------

__global__ void gat_edge_logit(const float* __restrict__ xl, const float* __restrict__ xr,
                               const float* __restrict__ att,
                               const int* __restrict__ src, const int* __restrict__ dst,
                               int E, int E2, int H, int C, int Dout,
                               float* __restrict__ elog, unsigned* __restrict__ mmax)
{
    int idx = blockIdx.x * blockDim.x + threadIdx.x;
    if (idx >= E2 * H) return;
    int e = idx / H, hh = idx - e * H;
    int s = (e < E) ? src[e] : (e - E);     // self loops appended
    int d = (e < E) ? dst[e] : (e - E);
    const float* pl = xl + (size_t)s * Dout + hh * C;
    const float* pr = xr + (size_t)d * Dout + hh * C;
    const float* pa = att + hh * C;
    float acc = 0.f;
#pragma unroll 8
    for (int c = 0; c < 64; ++c) {
        float v = pl[c] + pr[c];
        v = (v > 0.f) ? v : NEG_SLOPE * v;
        acc += v * pa[c];
    }
    elog[idx] = acc;
    atomicMax(&mmax[(size_t)d * H + hh], f2ord(acc));
}

__global__ void gat_edge_exp(float* __restrict__ elog, const unsigned* __restrict__ mmax,
                             float* __restrict__ denom, const int* __restrict__ dst,
                             int E, int E2, int H)
{
    int idx = blockIdx.x * blockDim.x + threadIdx.x;
    if (idx >= E2 * H) return;
    int e = idx / H, hh = idx - e * H;
    int d = (e < E) ? dst[e] : (e - E);
    float m = ord2f(mmax[(size_t)d * H + hh]);
    float ex = expf(elog[idx] - m);
    elog[idx] = ex;
    atomicAdd(&denom[(size_t)d * H + hh], ex);
}

__global__ void gat_edge_accum(const float* __restrict__ elog, const float* __restrict__ denom,
                               const float* __restrict__ xl,
                               const int* __restrict__ src, const int* __restrict__ dst,
                               int E, int E2, int H, int C, int Dout,
                               float* __restrict__ outb)
{
    int idx = blockIdx.x * blockDim.x + threadIdx.x;
    if (idx >= E2 * H) return;
    int e = idx / H, hh = idx - e * H;
    int s = (e < E) ? src[e] : (e - E);
    int d = (e < E) ? dst[e] : (e - E);
    float alpha = elog[idx] / (denom[(size_t)d * H + hh] + 1e-16f);
    const float* pl = xl + (size_t)s * Dout + hh * C;
    float* po = outb + (size_t)d * Dout + hh * C;
#pragma unroll 8
    for (int c = 0; c < 64; ++c) atomicAdd(&po[c], alpha * pl[c]);
}

// ------------------------------ GraphNorm ----------------------------------
// Segment sums with sorted-batch local accumulation (flush on graph change)
// to keep global atomic count low.

__global__ void seg_sum_kernel(const float* __restrict__ x, const int* __restrict__ batch,
                               float* __restrict__ out, int N, int D,
                               int outStride, int col0, int chunk,
                               const float* __restrict__ mean,
                               const float* __restrict__ ms, int sq)
{
    int d = blockIdx.x * blockDim.x + threadIdx.x;
    if (d >= D) return;
    int n0 = blockIdx.y * chunk;
    int n1 = n0 + chunk; if (n1 > N) n1 = N;
    float acc = 0.f;
    int curg = batch[n0];
    for (int n = n0; n < n1; ++n) {
        int g = batch[n];
        if (g != curg) {
            atomicAdd(&out[(size_t)curg * outStride + col0 + d], acc);
            acc = 0.f; curg = g;
        }
        float v = x[(size_t)n * D + d];
        if (sq) {
            float xc = v - ms[d] * mean[(size_t)g * D + d];
            v = xc * xc;
        }
        acc += v;
    }
    atomicAdd(&out[(size_t)curg * outStride + col0 + d], acc);
}

__global__ void div_cnt_kernel(float* buf, const unsigned* __restrict__ cnt, int G, int D) {
    int i = blockIdx.x * blockDim.x + threadIdx.x;
    if (i >= G * D) return;
    unsigned c = cnt[i / D];
    buf[i] /= (float)(c ? c : 1u);
}

__global__ void norm_elu_kernel(const float* __restrict__ x, const int* __restrict__ batch,
                                const float* __restrict__ mean, const float* __restrict__ var,
                                const float* __restrict__ w, const float* __restrict__ b,
                                const float* __restrict__ ms,
                                float* __restrict__ y, int N, int D)
{
    int idx = blockIdx.x * blockDim.x + threadIdx.x;
    if (idx >= N * D) return;
    int n = idx / D, d = idx - n * D;
    int g = batch[n];
    float xc = x[idx] - ms[d] * mean[(size_t)g * D + d];
    float v  = w[d] * xc * rsqrtf(var[(size_t)g * D + d] + GN_EPS) + b[d];
    y[idx] = (v > 0.f) ? v : expm1f(v);
}

// ------------------------------ final classifier ---------------------------

__global__ void pool_lin_kernel(const float* __restrict__ pooled,
                                const unsigned* __restrict__ cntg,
                                const float* __restrict__ linW, const float* __restrict__ linb,
                                float* __restrict__ out)
{
    __shared__ float sb[256];
    int g = blockIdx.x, tx = threadIdx.x;
    unsigned c = cntg[g];
    float cf = (float)(c ? c : 1u);
    float acc = 0.f;
    for (int j = tx; j < 1088; j += 256)
        acc += (pooled[g * 1088 + j] / cf) * linW[j];
    sb[tx] = acc;
    __syncthreads();
    for (int s = 128; s > 0; s >>= 1) {
        if (tx < s) sb[tx] += sb[tx + s];
        __syncthreads();
    }
    if (tx == 0) out[g] = sb[0] + linb[0];
}

// ------------------------------- host side ---------------------------------

struct LstmDir { const float* Wih; const float* Whh; const float* b; };

static inline int cdiv(int a, int b) { return (a + b - 1) / b; }

static void launch_gemm(hipStream_t s, const float* A, const float* W, const float* bias,
                        float* C, int M, int N, int K)
{
    int tiles = (M / 16) * (N / 16);
    gemm_wmma_bias<<<cdiv(tiles * 32, 256), 256, 0, s>>>(A, W, bias, C, M, N, K);
}

static void run_bilstm(hipStream_t s, const float* x0, int B, int T, int in0,
                       const LstmDir dirs[2][2], const float* fcW, const float* fcb,
                       const int* lens, float* ycat, float* gates,
                       float* h_f, float* c_f, float* h_b, float* c_b, float* hlog)
{
    for (int layer = 0; layer < 2; ++layer) {
        const float* in = (layer == 0) ? x0 : ycat;
        int in_d = (layer == 0) ? in0 : 128;
        float* yout = (layer == 0) ? ycat : nullptr;
        int gblocks = cdiv((B / 16) * 16 * 32, 256);
        int ublocks = cdiv(B * 64, 256);
        for (int dir = 0; dir < 2; ++dir) {
            const LstmDir& w = dirs[layer][dir];
            float* h = dir ? h_b : h_f;
            float* c = dir ? c_b : c_f;
            for (int t = 0; t < T; ++t) {
                lstm_gates_wmma<<<gblocks, 256, 0, s>>>(in, T, in_d, lens, t,
                    (t == 0) ? nullptr : h, w.Wih, w.Whh, w.b, gates, B, dir);
                lstm_update<<<ublocks, 256, 0, s>>>(gates,
                    (t == 0) ? nullptr : c, (t == 0) ? nullptr : h,
                    lens, t, h, c, yout, T, dir, B);
            }
        }
    }
    concat_h_kernel<<<cdiv(B * 128, 256), 256, 0, s>>>(h_f, h_b, gates, B);
    launch_gemm(s, gates, fcW, fcb, hlog, B, 64, 128);
}

static void gat_conv(hipStream_t s, const float* xin, int Din, int H, int C,
                     const float* Wl, const float* bl, const float* Wr, const float* br,
                     const float* att, const float* bias,
                     const int* src, const int* dst, int E, int N,
                     float* xl, float* xr, float* outb,
                     float* elog, unsigned* mmax, float* denom)
{
    int Dout = H * C;
    launch_gemm(s, xin, Wl, bl, xl, N, Dout, Din);
    launch_gemm(s, xin, Wr, br, xr, N, Dout, Din);
    zero_u32_kernel<<<cdiv(N * H, 256), 256, 0, s>>>(mmax, N * H);
    zero_f32_kernel<<<cdiv(N * H, 256), 256, 0, s>>>(denom, N * H);
    zero_f32_kernel<<<cdiv(N * Dout, 256), 256, 0, s>>>(outb, N * Dout);
    int E2 = E + N;
    int nb = cdiv(E2 * H, 256);
    gat_edge_logit<<<nb, 256, 0, s>>>(xl, xr, att, src, dst, E, E2, H, C, Dout, elog, mmax);
    gat_edge_exp  <<<nb, 256, 0, s>>>(elog, mmax, denom, dst, E, E2, H);
    gat_edge_accum<<<nb, 256, 0, s>>>(elog, denom, xl, src, dst, E, E2, H, C, Dout, outb);
    bias_add_kernel<<<cdiv(N * Dout, 256), 256, 0, s>>>(outb, bias, N * Dout, Dout);
}

static void graph_norm_elu(hipStream_t s, const float* x, const int* batch,
                           const unsigned* cntg, const float* w, const float* b,
                           const float* ms, float* meanb, float* varb,
                           float* y, int N, int D)
{
    int zb = cdiv(NG * D, 256);
    dim3 gs(cdiv(D, 256), cdiv(N, 64));
    zero_f32_kernel<<<zb, 256, 0, s>>>(meanb, NG * D);
    seg_sum_kernel<<<gs, 256, 0, s>>>(x, batch, meanb, N, D, D, 0, 64, nullptr, nullptr, 0);
    div_cnt_kernel<<<zb, 256, 0, s>>>(meanb, cntg, NG, D);
    zero_f32_kernel<<<zb, 256, 0, s>>>(varb, NG * D);
    seg_sum_kernel<<<gs, 256, 0, s>>>(x, batch, varb, N, D, D, 0, 64, meanb, ms, 1);
    div_cnt_kernel<<<zb, 256, 0, s>>>(varb, cntg, NG, D);
    norm_elu_kernel<<<cdiv(N * D, 256), 256, 0, s>>>(x, batch, meanb, varb, w, b, ms, y, N, D);
}

extern "C" void kernel_launch(void* const* d_in, const int* in_sizes, int n_in,
                              void* d_out, int out_size, void* d_ws, size_t ws_size,
                              hipStream_t stream)
{
#define F32(i) ((const float*)d_in[(i)])
#define I32(i) ((const int*)d_in[(i)])
    // Top-level inputs in setup_inputs() insertion order; nested 'params'
    // flattened with jax pytree convention (sorted dict keys, lists in order).
    const float* x_text   = F32(0);
    const float* x_vision = F32(1);
    const float* x_audio  = F32(2);
    // params: audio_lstm{fc_W,fc_b,layers[0]{bwd{W_hh,W_ih,b},fwd{...}},layers[1]{...}}
    LstmDir adirs[2][2] = {
        { { F32(9),  F32(8),  F32(10) },   // layer0 fwd  (W_ih, W_hh, b)
          { F32(6),  F32(5),  F32(7)  } }, // layer0 bwd
        { { F32(15), F32(14), F32(16) },
          { F32(12), F32(11), F32(13) } },
    };
    const float* a_fcW = F32(3);  const float* a_fcb = F32(4);
    // conv1..conv4: {Wl, Wr, att, bias, bl, br}
    const float *c1Wl = F32(17), *c1Wr = F32(18), *c1att = F32(19),
                *c1bias = F32(20), *c1bl = F32(21), *c1br = F32(22);
    const float *c2Wl = F32(23), *c2Wr = F32(24), *c2att = F32(25),
                *c2bias = F32(26), *c2bl = F32(27), *c2br = F32(28);
    const float *c4Wl = F32(29), *c4Wr = F32(30), *c4att = F32(31),
                *c4bias = F32(32), *c4bl = F32(33), *c4br = F32(34);
    const float *linW = F32(35), *linb = F32(36);
    // norms: {bias, mean_scale, weight}
    const float *n1b = F32(37), *n1ms = F32(38), *n1w = F32(39);
    const float *n2b = F32(40), *n2ms = F32(41), *n2w = F32(42);
    const float *n4b = F32(43), *n4ms = F32(44), *n4w = F32(45);
    const float *tpW = F32(46), *tpb = F32(47);
    const float *v_fcW = F32(48), *v_fcb = F32(49);
    LstmDir vdirs[2][2] = {
        { { F32(54), F32(53), F32(55) },
          { F32(51), F32(50), F32(52) } },
        { { F32(60), F32(59), F32(61) },
          { F32(57), F32(56), F32(58) } },
    };
    const int* edge_index = I32(62);
    const int* batch      = I32(63);
    const int* vlens      = I32(64);
    const int* vmap       = I32(65);
    const int* vidx       = I32(66);
    const int* alens      = I32(67);
    const int* amap       = I32(68);
    const int* aidx       = I32(69);
    const int* esrc = edge_index;
    const int* edst = edge_index + NE;
    float* out = (float*)d_out;

    // ---------------- workspace carve-out ----------------
    char* base = (char*)d_ws;
    size_t off = 0;
    auto alloc = [&](size_t bytes) -> void* {
        off = (off + 255) & ~(size_t)255;
        void* r = base + off;
        off += bytes;
        return r;
    };
    float* fx   = (float*)alloc((size_t)NN * 64 * 4);     // xp -> final_x
    float* x1   = (float*)alloc((size_t)NN * 512 * 4);
    float* x2   = (float*)alloc((size_t)NN * 512 * 4);
    float* x3   = (float*)alloc((size_t)NN * 64 * 4);
    // big region: LSTM phase uses it as y_cat (B*T*128); GAT phase as xl|xr|out
    char*  big  = (char*)alloc((size_t)NAUD * TAUD * 128 * 4);   // 100663296 B
    float* ycat = (float*)big;
    float* xl   = (float*)big;
    float* xr   = xl + (size_t)NN * 512;
    float* outb = xr + (size_t)NN * 512;
    float* gates= (float*)alloc((size_t)4096 * 256 * 4);
    float* h_f  = (float*)alloc((size_t)4096 * 64 * 4);
    float* c_f  = (float*)alloc((size_t)4096 * 64 * 4);
    float* h_b  = (float*)alloc((size_t)4096 * 64 * 4);
    float* c_b  = (float*)alloc((size_t)4096 * 64 * 4);
    float* hlog = (float*)alloc((size_t)4096 * 64 * 4);
    float* msum = (float*)alloc((size_t)4096 * 64 * 4);
    float* mcnt = (float*)alloc((size_t)4096 * 4);
    float* elog = (float*)alloc((size_t)(NE + NN) * HEADS * 4);
    unsigned* mmax = (unsigned*)alloc((size_t)NN * HEADS * 4);
    float* denom= (float*)alloc((size_t)NN * HEADS * 4);
    float* meanb= (float*)alloc((size_t)NG * 512 * 4);
    float* varb = (float*)alloc((size_t)NG * 512 * 4);
    unsigned* cntg = (unsigned*)alloc((size_t)NG * 4);
    float* pooled  = (float*)alloc((size_t)NG * 1088 * 4);

    // ---------------- 1) text projection: xp = x_text @ W^T + b -----------
    launch_gemm(stream, x_text, tpW, tpb, fx, NN, 64, TEXTD);

    // ---------------- 2) vision BiLSTM + fusion ---------------------------
    run_bilstm(stream, x_vision, NVIS, TVIS, 256, vdirs, v_fcW, v_fcb, vlens,
               ycat, gates, h_f, c_f, h_b, c_b, hlog);
    zero_f32_kernel<<<cdiv(NVIS * 64, 256), 256, 0, stream>>>(msum, NVIS * 64);
    zero_f32_kernel<<<cdiv(NVIS, 256), 256, 0, stream>>>(mcnt, NVIS);
    mod_scatter_kernel<<<cdiv(NVIS * 64, 256), 256, 0, stream>>>(hlog, vmap, msum, mcnt, NVIS);
    mod_apply_kernel<<<NVIS, 64, 0, stream>>>(msum, mcnt, vidx, fx);

    // ---------------- 3) audio BiLSTM + fusion ----------------------------
    run_bilstm(stream, x_audio, NAUD, TAUD, 128, adirs, a_fcW, a_fcb, alens,
               ycat, gates, h_f, c_f, h_b, c_b, hlog);
    zero_f32_kernel<<<cdiv(NAUD * 64, 256), 256, 0, stream>>>(msum, NAUD * 64);
    zero_f32_kernel<<<cdiv(NAUD, 256), 256, 0, stream>>>(mcnt, NAUD);
    mod_scatter_kernel<<<cdiv(NAUD * 64, 256), 256, 0, stream>>>(hlog, amap, msum, mcnt, NAUD);
    mod_apply_kernel<<<NAUD, 64, 0, stream>>>(msum, mcnt, aidx, fx);

    // ---------------- 4) per-graph node counts ----------------------------
    zero_u32_kernel<<<1, NG, 0, stream>>>(cntg, NG);
    batch_count_kernel<<<cdiv(NN, 256), 256, 0, stream>>>(batch, cntg, NN);

    // ---------------- 5) GATv2 stack --------------------------------------
    gat_conv(stream, fx, 64, HEADS, HID, c1Wl, c1bl, c1Wr, c1br, c1att, c1bias,
             esrc, edst, NE, NN, xl, xr, outb, elog, mmax, denom);
    graph_norm_elu(stream, outb, batch, cntg, n1w, n1b, n1ms, meanb, varb, x1, NN, 512);

    gat_conv(stream, x1, 512, HEADS, HID, c2Wl, c2bl, c2Wr, c2br, c2att, c2bias,
             esrc, edst, NE, NN, xl, xr, outb, elog, mmax, denom);
    graph_norm_elu(stream, outb, batch, cntg, n2w, n2b, n2ms, meanb, varb, x2, NN, 512);

    gat_conv(stream, x2, 512, 1, HID, c4Wl, c4bl, c4Wr, c4br, c4att, c4bias,
             esrc, edst, NE, NN, xl, xr, outb, elog, mmax, denom);
    graph_norm_elu(stream, outb, batch, cntg, n4w, n4b, n4ms, meanb, varb, x3, NN, 64);

    // ---------------- 6) JK-cat mean-pool + linear head -------------------
    zero_f32_kernel<<<cdiv(NG * 1088, 256), 256, 0, stream>>>(pooled, NG * 1088);
    {
        dim3 g1(cdiv(512, 256), cdiv(NN, 64));
        seg_sum_kernel<<<g1, 256, 0, stream>>>(x1, batch, pooled, NN, 512, 1088, 0,    64, nullptr, nullptr, 0);
        seg_sum_kernel<<<g1, 256, 0, stream>>>(x2, batch, pooled, NN, 512, 1088, 512,  64, nullptr, nullptr, 0);
        dim3 g3(cdiv(64, 256), cdiv(NN, 64));
        seg_sum_kernel<<<g3, 256, 0, stream>>>(x3, batch, pooled, NN, 64,  1088, 1024, 64, nullptr, nullptr, 0);
    }
    pool_lin_kernel<<<NG, 256, 0, stream>>>(pooled, cntg, linW, linb, out);
#undef F32
#undef I32
}
